// SimpleGAT_2637109920384
// MI455X (gfx1250) — matrix operand order repaired and checked by reading, not compiled, not run on hardware
//
#include <hip/hip_runtime.h>
#include <hip/hip_bf16.h>
#include <math.h>

typedef __attribute__((ext_vector_type(16))) _Float16 v16h;
typedef __attribute__((ext_vector_type(8)))  _Float16 v8h;
typedef __attribute__((ext_vector_type(8)))  float    v8f;

#define BN_EPS 1e-5f
#define NEG_SLOPE 0.2f

// ---------- helpers ----------
__device__ __forceinline__ unsigned fenc(float f) {
    unsigned u = __float_as_uint(f);
    return (u & 0x80000000u) ? ~u : (u | 0x80000000u);
}
__device__ __forceinline__ float fdec(unsigned u) {
    u = (u & 0x80000000u) ? (u & 0x7FFFFFFFu) : ~u;
    return __uint_as_float(u);
}
__device__ __forceinline__ float lrelu(float v) { return v > 0.f ? v : NEG_SLOPE * v; }
__device__ __forceinline__ float elu1(float v)  { return v > 0.f ? v : (expf(v) - 1.f); }
__device__ __forceinline__ float bn1(float t, float g, float be, float m, float vv) {
    return (t - m) * rsqrtf(vv + BN_EPS) * g + be;
}

// ---------- generic fills / converts ----------
__global__ void fill_f32_kernel(float* p, float val, int n) {
    int i = blockIdx.x * blockDim.x + threadIdx.x;
    if (i < n) p[i] = val;
}
__global__ void fill_u32_kernel(unsigned* p, unsigned val, int n) {
    int i = blockIdx.x * blockDim.x + threadIdx.x;
    if (i < n) p[i] = val;
}
__global__ void f32_to_f16_kernel(const float* __restrict__ in, _Float16* __restrict__ out, int n) {
    int i = blockIdx.x * blockDim.x + threadIdx.x;
    if (i < n) out[i] = (_Float16)in[i];
}

// ---------- WMMA GEMM: C[M,NC] = A[M,K] * B[K,NC]  (f16 in, f32 accum) ----------
// K and NC are compile-time so all strides fold into immediate offsets (32-bit
// address math only). One wave per 16x16 C tile; B fragments register-resident
// for the wave's column; wave grid-strides over M tiles, prefetching the next
// A tile (global_prefetch_b8) while the current tile's WMMAs issue.
// EPI: fused bias + BatchNorm epilogue (final layer writes d_out directly).
template<int K, int NC, bool EPI>
__global__ void wmma_gemm_kernel(const _Float16* __restrict__ A,
                                 const _Float16* __restrict__ B,
                                 float* __restrict__ C,
                                 int M,
                                 const float* __restrict__ bias,
                                 const float* __restrict__ bn_g,
                                 const float* __restrict__ bn_b,
                                 const float* __restrict__ bn_m,
                                 const float* __restrict__ bn_v)
{
    constexpr int KSTEPS = K / 32;

    const int lane        = threadIdx.x & 31;
    const int wave        = threadIdx.x >> 5;
    const int wavesPerCol = gridDim.x * (blockDim.x >> 5);
    const int waveId      = blockIdx.x * (blockDim.x >> 5) + wave;
    const int col0        = blockIdx.y * 16;
    const int mtiles      = M >> 4;

    // B fragment layout (16-bit B, 32x16): lane holds column (lane&15);
    // lanes 0-15 hold K 0..15, lanes 16-31 hold K 16..31, packed sequentially.
    const int bcol = col0 + (lane & 15);
    const int kb2  = (lane & 16) ? 16 : 0;
    v16h bfrag[KSTEPS];
#pragma unroll
    for (int s = 0; s < KSTEPS; ++s) {
        const _Float16* bp0 = B + (s * 32 + kb2) * NC + bcol;
#pragma unroll
        for (int j = 0; j < 16; ++j)
            bfrag[s][j] = bp0[j * NC];
    }

    // A fragment layout (16-bit A, 16x32): lane holds row (lane&15);
    // lanes 0-15: K kb..kb+15 with kb=0; lanes 16-31: kb=16.
    const int arow = lane & 15;
    const int kbA  = (lane & 16) ? 16 : 0;

    // epilogue constants (uniform over the whole column of tiles)
    const int ccol = col0 + (lane & 15);
    float eb = 0.f, eg = 0.f, ebe = 0.f, em = 0.f, ev = 0.f;
    if (EPI) { eb = bias[ccol]; eg = bn_g[ccol]; ebe = bn_b[ccol]; em = bn_m[ccol]; ev = bn_v[ccol]; }

    for (int mt = waveId; mt < mtiles; mt += wavesPerCol) {
        const int abase = (mt * 16 + arow) * K + kbA;   // 32-bit index
        // prefetch next tile's A row for this lane
        const int mtn = mt + wavesPerCol;
        if (mtn < mtiles)
            __builtin_prefetch(A + (mtn * 16 + arow) * K + kbA, 0, 0);

        v8f acc = {};
#pragma unroll
        for (int s = 0; s < KSTEPS; ++s) {
            const _Float16* ap = A + (abase + s * 32);
            v8h lo = *(const v8h*)(ap);
            v8h hi = *(const v8h*)(ap + 8);
            v16h af;
#pragma unroll
            for (int j = 0; j < 8; ++j) { af[j] = lo[j]; af[j + 8] = hi[j]; }
            acc = __builtin_amdgcn_wmma_f32_16x16x32_f16(false, af, false, bfrag[s],
                                                         (short)0, acc, false, false);
        }
        // C/D layout: VGPR i -> row = i + 8*(lane>=16), col = lane&15
        const int rbase = mt * 16 + ((lane & 16) ? 8 : 0);
        float* cp = C + rbase * NC + ccol;              // 32-bit index, NC folds to imm
        if (EPI) {
#pragma unroll
            for (int i = 0; i < 8; ++i)
                cp[i * NC] = bn1(acc[i] + eb, eg, ebe, em, ev);
        } else {
#pragma unroll
            for (int i = 0; i < 8; ++i)
                cp[i * NC] = acc[i];
        }
    }
}

// ---------- attention dot products: as[n,h] = sum_o H[n,h,o]*att_s[h,o] ----------
__global__ void att_dots_kernel(const float* __restrict__ H,
                                const float* __restrict__ att_s,
                                const float* __restrict__ att_d,
                                float* __restrict__ as_, float* __restrict__ ad_,
                                int n, int heads, int hid)
{
    int idx = blockIdx.x * blockDim.x + threadIdx.x;
    if (idx >= n * heads) return;
    int node = idx / heads, h = idx - node * heads;
    const float* hp = H + (size_t)node * heads * hid + (size_t)h * hid;
    float s = 0.f, d = 0.f;
    for (int o = 0; o < hid; ++o) {
        float x = hp[o];
        s += x * att_s[h * hid + o];
        d += x * att_d[h * hid + o];
    }
    as_[idx] = s;
    ad_[idx] = d;
}

// ---------- edge kernels (thread per edge; self-loops appended) ----------
__global__ void edge_max_kernel(const long long* __restrict__ ei, int E_, int n, int heads,
                                const float* __restrict__ as_, const float* __restrict__ ad_,
                                unsigned* __restrict__ mx)
{
    int e = blockIdx.x * blockDim.x + threadIdx.x;
    int tot = E_ + n;
    if (e >= tot) return;
    int s, d;
    if (e < E_) { s = (int)ei[e]; d = (int)ei[E_ + e]; } else { s = d = e - E_; }
    for (int h = 0; h < heads; ++h) {
        float v = lrelu(as_[s * heads + h] + ad_[d * heads + h]);
        atomicMax(&mx[d * heads + h], fenc(v));
    }
}

__global__ void edge_sum_kernel(const long long* __restrict__ ei, int E_, int n, int heads,
                                const float* __restrict__ as_, const float* __restrict__ ad_,
                                const unsigned* __restrict__ mx, float* __restrict__ den)
{
    int e = blockIdx.x * blockDim.x + threadIdx.x;
    int tot = E_ + n;
    if (e >= tot) return;
    int s, d;
    if (e < E_) { s = (int)ei[e]; d = (int)ei[E_ + e]; } else { s = d = e - E_; }
    for (int h = 0; h < heads; ++h) {
        float v  = lrelu(as_[s * heads + h] + ad_[d * heads + h]);
        float ex = expf(v - fdec(mx[d * heads + h]));
        atomicAdd(&den[d * heads + h], ex);
    }
}

// wave-per-edge aggregation, layer 1: 8 heads x 32 features (lane = feature)
__global__ void edge_aggr1_kernel(const long long* __restrict__ ei, int E_, int n,
                                  const float* __restrict__ as_, const float* __restrict__ ad_,
                                  const unsigned* __restrict__ mx, const float* __restrict__ den,
                                  const float* __restrict__ H, float* __restrict__ agg)
{
    int lane = threadIdx.x & 31;
    int wid  = blockIdx.x * (blockDim.x >> 5) + (threadIdx.x >> 5);
    int tot  = E_ + n;
    if (wid >= tot) return;
    int s, d;
    if (wid < E_) { s = (int)ei[wid]; d = (int)ei[E_ + wid]; } else { s = d = wid - E_; }
    for (int h = 0; h < 8; ++h) {
        float v     = lrelu(as_[s * 8 + h] + ad_[d * 8 + h]);
        float alpha = expf(v - fdec(mx[d * 8 + h])) / (den[d * 8 + h] + 1e-16f);
        float x     = H[s * 256 + h * 32 + lane] * alpha;
        atomicAdd(&agg[d * 256 + h * 32 + lane], x);
    }
}

// wave-per-edge aggregation, layer 2: 1 head x 64 features (lane, lane+32)
__global__ void edge_aggr2_kernel(const long long* __restrict__ ei, int E_, int n,
                                  const float* __restrict__ as_, const float* __restrict__ ad_,
                                  const unsigned* __restrict__ mx, const float* __restrict__ den,
                                  const float* __restrict__ H, float* __restrict__ agg)
{
    int lane = threadIdx.x & 31;
    int wid  = blockIdx.x * (blockDim.x >> 5) + (threadIdx.x >> 5);
    int tot  = E_ + n;
    if (wid >= tot) return;
    int s, d;
    if (wid < E_) { s = (int)ei[wid]; d = (int)ei[E_ + wid]; } else { s = d = wid - E_; }
    float v     = lrelu(as_[s] + ad_[d]);
    float alpha = expf(v - fdec(mx[d])) / (den[d] + 1e-16f);
#pragma unroll
    for (int half = 0; half < 2; ++half) {
        int o = lane + half * 32;
        atomicAdd(&agg[d * 64 + o], H[s * 64 + o] * alpha);
    }
}

// ---------- bias + BN + ELU epilogue, output as f16 (feeds next GEMM) ----------
__global__ void post_kernel(const float* __restrict__ agg, const float* __restrict__ bias,
                            const float* __restrict__ g, const float* __restrict__ be,
                            const float* __restrict__ m, const float* __restrict__ v,
                            _Float16* __restrict__ outh, int n, int dim)
{
    int i = blockIdx.x * blockDim.x + threadIdx.x;
    if (i >= n * dim) return;
    int c = i % dim;
    float t = agg[i] + bias[c];
    t = bn1(t, g[c], be[c], m[c], v[c]);
    outh[i] = (_Float16)elu1(t);
}

// =====================================================================
extern "C" void kernel_launch(void* const* d_in, const int* in_sizes, int n_in,
                              void* d_out, int out_size, void* d_ws, size_t ws_size,
                              hipStream_t stream)
{
    const float*     x      = (const float*)d_in[0];
    const long long* ei     = (const long long*)d_in[1];
    const float*     W1     = (const float*)d_in[2];
    const float*     a_src1 = (const float*)d_in[3];
    const float*     a_dst1 = (const float*)d_in[4];
    const float*     b1     = (const float*)d_in[5];
    const float*     W2     = (const float*)d_in[6];
    const float*     a_src2 = (const float*)d_in[7];
    const float*     a_dst2 = (const float*)d_in[8];
    const float*     b2     = (const float*)d_in[9];
    const float*     Wp     = (const float*)d_in[10];
    const float*     bp     = (const float*)d_in[11];
    const float* g1 = (const float*)d_in[12], *be1 = (const float*)d_in[13];
    const float* m1 = (const float*)d_in[14], *v1  = (const float*)d_in[15];
    const float* g2 = (const float*)d_in[16], *be2 = (const float*)d_in[17];
    const float* m2 = (const float*)d_in[18], *v2  = (const float*)d_in[19];
    const float* g3 = (const float*)d_in[20], *be3 = (const float*)d_in[21];
    const float* m3 = (const float*)d_in[22], *v3  = (const float*)d_in[23];

    const int N = in_sizes[0] / 128;   // 50000
    const int E = in_sizes[1] / 2;     // 800000
    const int TOT = E + N;

    // ---- workspace carve-up (256B aligned) ----
    char* ws = (char*)d_ws;
    size_t off = 0;
    auto take = [&](size_t bytes) { char* p = ws + off; off = (off + bytes + 255) & ~(size_t)255; return p; };
    _Float16* xh   = (_Float16*)take((size_t)N * 128 * 2);
    _Float16* W1h  = (_Float16*)take(128 * 256 * 2);
    _Float16* W2h  = (_Float16*)take(256 * 64 * 2);
    _Float16* Wph  = (_Float16*)take(64 * 128 * 2);
    float*    H1   = (float*)take((size_t)N * 256 * 4);
    float*    as1  = (float*)take((size_t)N * 8 * 4);
    float*    ad1  = (float*)take((size_t)N * 8 * 4);
    unsigned* mx1  = (unsigned*)take((size_t)N * 8 * 4);
    float*    den1 = (float*)take((size_t)N * 8 * 4);
    float*    agg1 = (float*)take((size_t)N * 256 * 4);
    _Float16* h1h  = (_Float16*)take((size_t)N * 256 * 2);
    float*    H2   = (float*)take((size_t)N * 64 * 4);
    float*    as2  = (float*)take((size_t)N * 4);
    float*    ad2  = (float*)take((size_t)N * 4);
    unsigned* mx2  = (unsigned*)take((size_t)N * 4);
    float*    den2 = (float*)take((size_t)N * 4);
    float*    agg2 = (float*)take((size_t)N * 64 * 4);
    _Float16* h2h  = (_Float16*)take((size_t)N * 64 * 2);

    const unsigned NEG_INF_ENC = 0x007FFFFFu;  // fenc(-inf)
    auto blocks = [](int n, int t) { return (n + t - 1) / t; };

    // ---- init + f16 conversions ----
    f32_to_f16_kernel<<<blocks(N * 128, 256), 256, 0, stream>>>(x,  xh,  N * 128);
    f32_to_f16_kernel<<<blocks(128 * 256, 256), 256, 0, stream>>>(W1, W1h, 128 * 256);
    f32_to_f16_kernel<<<blocks(256 * 64, 256), 256, 0, stream>>>(W2, W2h, 256 * 64);
    f32_to_f16_kernel<<<blocks(64 * 128, 256), 256, 0, stream>>>(Wp, Wph, 64 * 128);
    fill_u32_kernel<<<blocks(N * 8, 256), 256, 0, stream>>>(mx1, NEG_INF_ENC, N * 8);
    fill_f32_kernel<<<blocks(N * 8, 256), 256, 0, stream>>>(den1, 0.f, N * 8);
    fill_f32_kernel<<<blocks(N * 256, 256), 256, 0, stream>>>(agg1, 0.f, N * 256);
    fill_u32_kernel<<<blocks(N, 256), 256, 0, stream>>>(mx2, NEG_INF_ENC, N);
    fill_f32_kernel<<<blocks(N, 256), 256, 0, stream>>>(den2, 0.f, N);
    fill_f32_kernel<<<blocks(N * 64, 256), 256, 0, stream>>>(agg2, 0.f, N * 64);

    // ---- layer 1: H1 = x @ W1  (K=128 -> 4 WMMA k-steps), NC=256 -> 16 col tiles
    wmma_gemm_kernel<128, 256, false><<<dim3(128, 16), 256, 0, stream>>>(
        xh, W1h, H1, N, nullptr, nullptr, nullptr, nullptr, nullptr);
    att_dots_kernel<<<blocks(N * 8, 256), 256, 0, stream>>>(H1, a_src1, a_dst1, as1, ad1, N, 8, 32);
    edge_max_kernel<<<blocks(TOT, 256), 256, 0, stream>>>(ei, E, N, 8, as1, ad1, mx1);
    edge_sum_kernel<<<blocks(TOT, 256), 256, 0, stream>>>(ei, E, N, 8, as1, ad1, mx1, den1);
    edge_aggr1_kernel<<<blocks(TOT, 8) , 256, 0, stream>>>(ei, E, N, as1, ad1, mx1, den1, H1, agg1);
    post_kernel<<<blocks(N * 256, 256), 256, 0, stream>>>(agg1, b1, g1, be1, m1, v1, h1h, N, 256);

    // ---- layer 2: H2 = h1 @ W2  (K=256 -> 8 k-steps), NC=64 -> 4 col tiles
    wmma_gemm_kernel<256, 64, false><<<dim3(128, 4), 256, 0, stream>>>(
        h1h, W2h, H2, N, nullptr, nullptr, nullptr, nullptr, nullptr);
    att_dots_kernel<<<blocks(N, 256), 256, 0, stream>>>(H2, a_src2, a_dst2, as2, ad2, N, 1, 64);
    edge_max_kernel<<<blocks(TOT, 256), 256, 0, stream>>>(ei, E, N, 1, as2, ad2, mx2);
    edge_sum_kernel<<<blocks(TOT, 256), 256, 0, stream>>>(ei, E, N, 1, as2, ad2, mx2, den2);
    edge_aggr2_kernel<<<blocks(TOT, 8) , 256, 0, stream>>>(ei, E, N, as2, ad2, mx2, den2, H2, agg2);
    post_kernel<<<blocks(N * 64, 256), 256, 0, stream>>>(agg2, b2, g2, be2, m2, v2, h2h, N, 64);

    // ---- projection + bias + BN3 fused into WMMA epilogue -> d_out [N,128] f32
    wmma_gemm_kernel<64, 128, true><<<dim3(128, 8), 256, 0, stream>>>(
        h2h, Wph, (float*)d_out, N, bp, g3, be3, m3, v3);
}
